// TransformerBlock_49890340111095
// MI455X (gfx1250) — matrix operand order repaired and checked by reading, not compiled, not run on hardware
//
#include <hip/hip_runtime.h>
#include <hip/hip_bf16.h>
#include <math.h>

// ---------------- problem constants ----------------
#define Bk     2
#define Sk     2048
#define Dk     2048
#define Hk     16
#define NOPEk  128
#define ROPEk  64
#define VDk    128
#define QLORA  1536
#define KVLORA 512
#define Ek     8
#define Mk     1408
#define SHMk   5632
#define Tk     (Bk*Sk)                 // 4096 tokens
#define ROWS_TOT (Tk*2 + Ek*64)        // 8704: top-2 assignments, 64-aligned per-expert segments

// ---------------- WMMA types / helpers ----------------
typedef __attribute__((ext_vector_type(16))) __bf16 v16bf;
typedef __attribute__((ext_vector_type(8)))  float  v8f;

union FragBF { unsigned short u[16]; unsigned int w[8]; v16bf v; };

__device__ __forceinline__ v8f wmma_bf16(v16bf a, v16bf b, v8f c) {
  return __builtin_amdgcn_wmma_f32_16x16x32_bf16(false, a, false, b, (short)0, c, false, false);
}
__device__ __forceinline__ unsigned short f2bf(float f) {
  unsigned int u = __float_as_uint(f);
  return (unsigned short)((u + 0x7FFFu + ((u >> 16) & 1u)) >> 16);
}
__device__ __forceinline__ float bf2f(unsigned short h) {
  return __uint_as_float(((unsigned int)h) << 16);
}

// ---------------- rmsnorm ----------------
__global__ __launch_bounds__(256)
void rmsnorm_f32_k(const float* __restrict__ x, const float* __restrict__ w,
                   unsigned short* __restrict__ out, int D) {
  __shared__ float red[256];
  int t = blockIdx.x, tid = threadIdx.x;
  const float* row = x + (size_t)t * D;
  float acc = 0.f;
  for (int d = tid; d < D; d += 256) { float v = row[d]; acc += v * v; }
  red[tid] = acc; __syncthreads();
  for (int s = 128; s > 0; s >>= 1) { if (tid < s) red[tid] += red[tid + s]; __syncthreads(); }
  float inv = rsqrtf(red[0] / (float)D + 1e-6f);
  for (int d = tid; d < D; d += 256) out[(size_t)t * D + d] = f2bf(row[d] * inv * w[d]);
}

__global__ __launch_bounds__(256)
void rmsnorm_bf16_k(unsigned short* __restrict__ x, int ld, int W, const float* __restrict__ w) {
  __shared__ float red[256];
  int t = blockIdx.x, tid = threadIdx.x;
  unsigned short* row = x + (size_t)t * ld;
  float acc = 0.f;
  for (int d = tid; d < W; d += 256) { float v = bf2f(row[d]); acc += v * v; }
  red[tid] = acc; __syncthreads();
  for (int s = 128; s > 0; s >>= 1) { if (tid < s) red[tid] += red[tid + s]; __syncthreads(); }
  float inv = rsqrtf(red[0] / (float)W + 1e-6f);
  for (int d = tid; d < W; d += 256) row[d] = f2bf(bf2f(row[d]) * inv * w[d]);
}

// ---------------- rope (in-place on bf16) ----------------
__global__ __launch_bounds__(256)
void rope_k(unsigned short* __restrict__ p, int ld, int nh, int hstride, int roff,
            const float* __restrict__ freqs, int T, int S) {
  int idx = blockIdx.x * blockDim.x + threadIdx.x;
  int total = T * nh * (ROPEk / 2);
  if (idx >= total) return;
  int i = idx & 31; int rest = idx >> 5;
  int h = rest % nh; int t = rest / nh;
  int s = t % S;
  float c  = freqs[((size_t)s * 32 + i) * 2 + 0];
  float sn = freqs[((size_t)s * 32 + i) * 2 + 1];
  unsigned short* base = p + (size_t)t * ld + h * hstride + roff + 2 * i;
  float x1 = bf2f(base[0]), x2 = bf2f(base[1]);
  base[0] = f2bf(x1 * c - x2 * sn);
  base[1] = f2bf(x1 * sn + x2 * c);
}

// ---------------- generic bf16 WMMA GEMM: C[T,N] = A[T,K](bf16) * W[N,K](f32)^T ----------------
// MODE 0: write f32; MODE 1: write bf16; MODE 2: write f32 + residual
template<int MODE>
__global__ __launch_bounds__(128)
void gemm_k(const unsigned short* __restrict__ A, int lda,
            const float* __restrict__ W,
            void* __restrict__ Cout, const float* __restrict__ Res,
            int N, int K) {
  __shared__ unsigned short As[64][32];
  __shared__ unsigned short Ws[64][32];
  int tid = threadIdx.x, wave = tid >> 5, lane = tid & 31;
  int half = lane >> 4, l16 = lane & 15;
  int tm = blockIdx.x * 64, tn = blockIdx.y * 64;
  int wm = (wave >> 1) * 32, wn = (wave & 1) * 32;
  v8f c[2][2] = {};
  for (int k0 = 0; k0 < K; k0 += 32) {
    __syncthreads();
    #pragma unroll
    for (int i = 0; i < 2; ++i) {
      int idx = tid + i * 128; int r = idx >> 2; int cc = (idx & 3) * 8;
      *(uint4*)&As[r][cc] = *(const uint4*)(A + (size_t)(tm + r) * lda + k0 + cc);
    }
    #pragma unroll
    for (int i = 0; i < 4; ++i) {
      int idx = tid + i * 128; int r = idx >> 3; int cc = (idx & 7) * 4;
      float4 f = *(const float4*)(W + (size_t)(tn + r) * K + k0 + cc);
      unsigned short* d = &Ws[r][cc];
      d[0] = f2bf(f.x); d[1] = f2bf(f.y); d[2] = f2bf(f.z); d[3] = f2bf(f.w);
    }
    __syncthreads();
    FragBF a[2], b[2];
    #pragma unroll
    for (int r = 0; r < 2; ++r) {
      int row = wm + r * 16 + l16;
      #pragma unroll
      for (int p = 0; p < 8; ++p) {
        int kk = 16 * half + 2 * p;                           // contiguous k layout (matches B)
        a[r].w[p] = *(const unsigned int*)&As[row][kk];
      }
    }
    #pragma unroll
    for (int ci = 0; ci < 2; ++ci) {
      int col = wn + ci * 16 + l16;
      #pragma unroll
      for (int p = 0; p < 8; ++p) {
        int kk = 16 * half + 2 * p;                           // contiguous k layout (matches A)
        b[ci].w[p] = *(const unsigned int*)&Ws[col][kk];
      }
    }
    #pragma unroll
    for (int r = 0; r < 2; ++r)
      #pragma unroll
      for (int ci = 0; ci < 2; ++ci)
        c[r][ci] = wmma_bf16(a[r].v, b[ci].v, c[r][ci]);
  }
  #pragma unroll
  for (int r = 0; r < 2; ++r)
    #pragma unroll
    for (int ci = 0; ci < 2; ++ci) {
      int gn = tn + wn + ci * 16 + l16;
      #pragma unroll
      for (int j = 0; j < 8; ++j) {
        int gm = tm + wm + r * 16 + j + 8 * half;
        float val = c[r][ci][j];
        size_t off = (size_t)gm * N + gn;
        if (MODE == 0)      ((float*)Cout)[off] = val;
        else if (MODE == 1) ((unsigned short*)Cout)[off] = f2bf(val);
        else                ((float*)Cout)[off] = val + Res[off];
      }
    }
}

// ---------------- flash attention (bf16 WMMA, online softmax) ----------------
__global__ __launch_bounds__(128)
void attention_k(const unsigned short* __restrict__ Q,    // [T, H*192]
                 const unsigned short* __restrict__ KV,   // [T, H*256] (128 k_nope | 128 v)
                 const unsigned short* __restrict__ KPE,  // row t at KPE + t*576 (64 elems)
                 unsigned short* __restrict__ O,          // [T, H*128]
                 int S) {
  __shared__ unsigned short Qs[64][192];
  __shared__ unsigned short Ks[32][192];
  __shared__ unsigned short Vst[128][32];                  // transposed V: [vdim][key]
  __shared__ unsigned short Ps[4][16][32];
  int tid = threadIdx.x, wave = tid >> 5, lane = tid & 31;
  int half = lane >> 4, l16 = lane & 15;
  int qb = blockIdx.x * 64, h = blockIdx.y, bz = blockIdx.z;
  size_t tbase = (size_t)bz * S;

  for (int idx = tid; idx < 64 * 24; idx += 128) {
    int r = idx / 24, cc = (idx % 24) * 8;
    *(uint4*)&Qs[r][cc] = *(const uint4*)(Q + (tbase + qb + r) * (Hk * 192) + h * 192 + cc);
  }

  v8f o[8]; float mrun[8], lrun[8];
  v8f zero = {};
  #pragma unroll
  for (int i = 0; i < 8; ++i) { o[i] = zero; mrun[i] = -1e30f; lrun[i] = 0.f; }
  const float scale = 0.0721687836f;  // 1/sqrt(192)

  int nt = (qb >> 5) + 2;
  for (int jt = 0; jt < nt; ++jt) {
    int kb = jt * 32;
    __syncthreads();
    for (int idx = tid; idx < 32 * 16; idx += 128) {        // k_nope
      int r = idx / 16, cc = (idx % 16) * 8;
      *(uint4*)&Ks[r][cc] = *(const uint4*)(KV + (tbase + kb + r) * (Hk * 256) + h * 256 + cc);
    }
    for (int idx = tid; idx < 32 * 8; idx += 128) {         // k_pe (shared across heads)
      int r = idx / 8, cc = (idx % 8) * 8;
      *(uint4*)&Ks[r][128 + cc] = *(const uint4*)(KPE + (tbase + kb + r) * 576 + cc);
    }
    for (int idx = tid; idx < 32 * 16; idx += 128) {        // V -> transposed LDS
      int r = idx / 16, cc = (idx % 16) * 8;
      uint4 pack = *(const uint4*)(KV + (tbase + kb + r) * (Hk * 256) + h * 256 + 128 + cc);
      const unsigned short* pv = (const unsigned short*)&pack;
      #pragma unroll
      for (int i = 0; i < 8; ++i) Vst[cc + i][r] = pv[i];
    }
    __syncthreads();

    v8f s0 = zero, s1 = zero;
    int qrow = wave * 16 + l16;
    #pragma unroll
    for (int kc = 0; kc < 6; ++kc) {
      FragBF a, b0, b1;
      #pragma unroll
      for (int p = 0; p < 8; ++p) {
        int kk = kc * 32 + 16 * half + 2 * p;
        a.w[p] = *(const unsigned int*)&Qs[qrow][kk];
      }
      #pragma unroll
      for (int p = 0; p < 8; ++p) {
        int kk = kc * 32 + 16 * half + 2 * p;
        b0.w[p] = *(const unsigned int*)&Ks[l16][kk];
        b1.w[p] = *(const unsigned int*)&Ks[l16 + 16][kk];
      }
      s0 = wmma_bf16(a.v, b0.v, s0);
      s1 = wmma_bf16(a.v, b1.v, s1);
    }

    float p0[8], p1[8], alpha[8];
    #pragma unroll
    for (int j = 0; j < 8; ++j) {
      int grow = qb + wave * 16 + j + 8 * half;
      float v0 = s0[j] * scale, v1 = s1[j] * scale;
      if (kb + l16 > grow)      v0 = -1e30f;                 // causal mask
      if (kb + 16 + l16 > grow) v1 = -1e30f;
      float cand = fmaxf(v0, v1);
      #pragma unroll
      for (int msk = 8; msk >= 1; msk >>= 1) cand = fmaxf(cand, __shfl_xor(cand, msk, 32));
      float mnew = fmaxf(mrun[j], cand);
      p0[j] = __expf(v0 - mnew);
      p1[j] = __expf(v1 - mnew);
      float rs = p0[j] + p1[j];
      #pragma unroll
      for (int msk = 8; msk >= 1; msk >>= 1) rs += __shfl_xor(rs, msk, 32);
      alpha[j] = __expf(mrun[j] - mnew);
      lrun[j] = lrun[j] * alpha[j] + rs;
      mrun[j] = mnew;
    }
    #pragma unroll
    for (int j = 0; j < 8; ++j) {                            // P -> per-wave LDS (transpose)
      int m = j + 8 * half;
      Ps[wave][m][l16]      = f2bf(p0[j]);
      Ps[wave][m][l16 + 16] = f2bf(p1[j]);
    }
    #pragma unroll
    for (int vc = 0; vc < 8; ++vc) {
      #pragma unroll
      for (int j = 0; j < 8; ++j) o[vc][j] *= alpha[j];
    }

    FragBF pa;
    #pragma unroll
    for (int p = 0; p < 8; ++p) {
      int kk = 16 * half + 2 * p;
      pa.w[p] = *(const unsigned int*)&Ps[wave][l16][kk];
    }
    #pragma unroll
    for (int vc = 0; vc < 8; ++vc) {
      FragBF bv;
      #pragma unroll
      for (int p = 0; p < 8; ++p) {
        int kk = 16 * half + 2 * p;
        bv.w[p] = *(const unsigned int*)&Vst[vc * 16 + l16][kk];
      }
      o[vc] = wmma_bf16(pa.v, bv.v, o[vc]);
    }
  }
  #pragma unroll
  for (int vc = 0; vc < 8; ++vc) {
    #pragma unroll
    for (int j = 0; j < 8; ++j) {
      int grow = qb + wave * 16 + j + 8 * half;
      float val = o[vc][j] / lrun[j];
      O[(tbase + grow) * (Hk * 128) + h * 128 + vc * 16 + l16] = f2bf(val);
    }
  }
}

// ---------------- router & MoE plumbing ----------------
__global__ void zero_counts_k(int* counts) { if (threadIdx.x < Ek) counts[threadIdx.x] = 0; }

__global__ __launch_bounds__(128)
void router_k(const unsigned short* __restrict__ X, const float* __restrict__ RW,
              int* __restrict__ topi, float* __restrict__ topv, int* __restrict__ counts) {
  __shared__ float red[Ek][128];
  int t = blockIdx.x, tid = threadIdx.x;
  float acc[Ek] = {};
  for (int d = tid; d < Dk; d += 128) {
    float xv = bf2f(X[(size_t)t * Dk + d]);
    #pragma unroll
    for (int e = 0; e < Ek; ++e) acc[e] += xv * RW[(size_t)e * Dk + d];
  }
  #pragma unroll
  for (int e = 0; e < Ek; ++e) red[e][tid] = acc[e];
  __syncthreads();
  for (int s = 64; s > 0; s >>= 1) {
    if (tid < s) {
      for (int e = 0; e < Ek; ++e) red[e][tid] += red[e][tid + s];
    }
    __syncthreads();
  }
  if (tid == 0) {
    float lg[Ek], mx = -1e30f;
    for (int e = 0; e < Ek; ++e) { lg[e] = red[e][0]; mx = fmaxf(mx, lg[e]); }
    float sum = 0.f;
    for (int e = 0; e < Ek; ++e) { lg[e] = __expf(lg[e] - mx); sum += lg[e]; }
    for (int e = 0; e < Ek; ++e) lg[e] /= sum;
    int i0 = 0; for (int e = 1; e < Ek; ++e) if (lg[e] > lg[i0]) i0 = e;
    int i1 = (i0 == 0) ? 1 : 0;
    for (int e = 0; e < Ek; ++e) if (e != i0 && lg[e] > lg[i1]) i1 = e;
    float v0 = lg[i0], v1 = lg[i1], s2 = v0 + v1;
    topi[t * 2] = i0; topi[t * 2 + 1] = i1;
    topv[t * 2] = v0 / s2; topv[t * 2 + 1] = v1 / s2;
    atomicAdd(&counts[i0], 1); atomicAdd(&counts[i1], 1);
  }
}

__global__ void seg_k(const int* __restrict__ counts, int* __restrict__ seg, int* __restrict__ pos) {
  if (threadIdx.x == 0) {
    seg[0] = 0;
    for (int e = 0; e < Ek; ++e) { seg[e + 1] = seg[e] + ((counts[e] + 63) & ~63); pos[e] = 0; }
  }
}

__global__ void assign_init_k(int* __restrict__ tok, float* __restrict__ gate) {
  int i = blockIdx.x * blockDim.x + threadIdx.x;
  if (i < ROWS_TOT) { tok[i] = 0; gate[i] = 0.f; }
}

__global__ void scatter_assign_k(const int* __restrict__ topi, const float* __restrict__ topv,
                                 const int* __restrict__ seg, int* __restrict__ pos,
                                 int* __restrict__ tok, float* __restrict__ gate) {
  int t = blockIdx.x * blockDim.x + threadIdx.x;
  if (t >= Tk) return;
  for (int j = 0; j < 2; ++j) {
    int e = topi[t * 2 + j];
    int p = atomicAdd(&pos[e], 1);
    int slot = seg[e] + p;
    tok[slot] = t; gate[slot] = topv[t * 2 + j];
  }
}

// ---------------- fused SiLU-gate FFN: out = silu(x*W1^T) * (x*W3^T), optional row gather ----------------
__global__ __launch_bounds__(128)
void ffn13_k(const unsigned short* __restrict__ X, int lda,
             const float* __restrict__ W1g, const float* __restrict__ W3g, size_t wstride,
             const int* __restrict__ gidx, const int* __restrict__ seg,
             unsigned short* __restrict__ Out, int ldo, int N, int K) {
  __shared__ unsigned short As[64][32];
  __shared__ unsigned short W1s[64][32];
  __shared__ unsigned short W3s[64][32];
  __shared__ int srow[64];
  int tid = threadIdx.x, wave = tid >> 5, lane = tid & 31;
  int half = lane >> 4, l16 = lane & 15;
  int rbase = blockIdx.x * 64, nbase = blockIdx.y * 64;
  int e = 0;
  if (seg) { while (e < Ek - 1 && rbase >= seg[e + 1]) ++e; }
  const float* W1 = W1g + (size_t)e * wstride;
  const float* W3 = W3g + (size_t)e * wstride;
  if (tid < 64) srow[tid] = gidx ? gidx[rbase + tid] : (rbase + tid);
  int wm = (wave >> 1) * 32, wn = (wave & 1) * 32;
  v8f c1[2][2] = {}; v8f c3[2][2] = {};
  for (int k0 = 0; k0 < K; k0 += 32) {
    __syncthreads();
    #pragma unroll
    for (int i = 0; i < 2; ++i) {
      int idx = tid + i * 128; int r = idx >> 2; int cc = (idx & 3) * 8;
      *(uint4*)&As[r][cc] = *(const uint4*)(X + (size_t)srow[r] * lda + k0 + cc);
    }
    #pragma unroll
    for (int i = 0; i < 4; ++i) {
      int idx = tid + i * 128; int r = idx >> 3; int cc = (idx & 7) * 4;
      float4 f1 = *(const float4*)(W1 + (size_t)(nbase + r) * K + k0 + cc);
      float4 f3 = *(const float4*)(W3 + (size_t)(nbase + r) * K + k0 + cc);
      unsigned short* d1 = &W1s[r][cc];
      d1[0] = f2bf(f1.x); d1[1] = f2bf(f1.y); d1[2] = f2bf(f1.z); d1[3] = f2bf(f1.w);
      unsigned short* d3 = &W3s[r][cc];
      d3[0] = f2bf(f3.x); d3[1] = f2bf(f3.y); d3[2] = f2bf(f3.z); d3[3] = f2bf(f3.w);
    }
    __syncthreads();
    FragBF a[2], b1[2], b3[2];
    #pragma unroll
    for (int r = 0; r < 2; ++r) {
      int row = wm + r * 16 + l16;
      #pragma unroll
      for (int p = 0; p < 8; ++p) {
        int kk = 16 * half + 2 * p;
        a[r].w[p] = *(const unsigned int*)&As[row][kk];
      }
    }
    #pragma unroll
    for (int ci = 0; ci < 2; ++ci) {
      int col = wn + ci * 16 + l16;
      #pragma unroll
      for (int p = 0; p < 8; ++p) {
        int kk = 16 * half + 2 * p;
        b1[ci].w[p] = *(const unsigned int*)&W1s[col][kk];
        b3[ci].w[p] = *(const unsigned int*)&W3s[col][kk];
      }
    }
    #pragma unroll
    for (int r = 0; r < 2; ++r) {
      #pragma unroll
      for (int ci = 0; ci < 2; ++ci) {
        c1[r][ci] = wmma_bf16(a[r].v, b1[ci].v, c1[r][ci]);
        c3[r][ci] = wmma_bf16(a[r].v, b3[ci].v, c3[r][ci]);
      }
    }
  }
  #pragma unroll
  for (int r = 0; r < 2; ++r) {
    #pragma unroll
    for (int ci = 0; ci < 2; ++ci) {
      int gn = nbase + wn + ci * 16 + l16;
      #pragma unroll
      for (int j = 0; j < 8; ++j) {
        int gm = rbase + wm + r * 16 + j + 8 * half;
        float a1 = c1[r][ci][j], a3 = c3[r][ci][j];
        float gA = a1 / (1.f + __expf(-a1));
        Out[(size_t)gm * ldo + gn] = f2bf(gA * a3);
      }
    }
  }
}

// ---------------- MoE second GEMM with gated scatter-accumulate ----------------
__global__ __launch_bounds__(128)
void moe_gemm2_k(const unsigned short* __restrict__ Hb,    // [ROWS_TOT, M] bf16
                 const float* __restrict__ W2g, size_t wstride,
                 const int* __restrict__ tok, const float* __restrict__ gate,
                 const int* __restrict__ seg,
                 float* __restrict__ Out, int N, int K) {
  __shared__ unsigned short As[64][32];
  __shared__ unsigned short Ws[64][32];
  __shared__ int   stok[64];
  __shared__ float sgate[64];
  int tid = threadIdx.x, wave = tid >> 5, lane = tid & 31;
  int half = lane >> 4, l16 = lane & 15;
  int rbase = blockIdx.x * 64, nbase = blockIdx.y * 64;
  int e = 0;
  while (e < Ek - 1 && rbase >= seg[e + 1]) ++e;
  const float* W2 = W2g + (size_t)e * wstride;
  if (tid < 64) { stok[tid] = tok[rbase + tid]; sgate[tid] = gate[rbase + tid]; }
  int wm = (wave >> 1) * 32, wn = (wave & 1) * 32;
  v8f c[2][2] = {};
  for (int k0 = 0; k0 < K; k0 += 32) {
    __syncthreads();
    #pragma unroll
    for (int i = 0; i < 2; ++i) {
      int idx = tid + i * 128; int r = idx >> 2; int cc = (idx & 3) * 8;
      *(uint4*)&As[r][cc] = *(const uint4*)(Hb + (size_t)(rbase + r) * K + k0 + cc);
    }
    #pragma unroll
    for (int i = 0; i < 4; ++i) {
      int idx = tid + i * 128; int r = idx >> 3; int cc = (idx & 7) * 4;
      float4 f = *(const float4*)(W2 + (size_t)(nbase + r) * K + k0 + cc);
      unsigned short* d = &Ws[r][cc];
      d[0] = f2bf(f.x); d[1] = f2bf(f.y); d[2] = f2bf(f.z); d[3] = f2bf(f.w);
    }
    __syncthreads();
    FragBF a[2], b[2];
    #pragma unroll
    for (int r = 0; r < 2; ++r) {
      int row = wm + r * 16 + l16;
      #pragma unroll
      for (int p = 0; p < 8; ++p) {
        int kk = 16 * half + 2 * p;
        a[r].w[p] = *(const unsigned int*)&As[row][kk];
      }
    }
    #pragma unroll
    for (int ci = 0; ci < 2; ++ci) {
      int col = wn + ci * 16 + l16;
      #pragma unroll
      for (int p = 0; p < 8; ++p) {
        int kk = 16 * half + 2 * p;
        b[ci].w[p] = *(const unsigned int*)&Ws[col][kk];
      }
    }
    #pragma unroll
    for (int r = 0; r < 2; ++r) {
      #pragma unroll
      for (int ci = 0; ci < 2; ++ci)
        c[r][ci] = wmma_bf16(a[r].v, b[ci].v, c[r][ci]);
    }
  }
  #pragma unroll
  for (int r = 0; r < 2; ++r) {
    #pragma unroll
    for (int ci = 0; ci < 2; ++ci) {
      int gn = nbase + wn + ci * 16 + l16;
      #pragma unroll
      for (int j = 0; j < 8; ++j) {
        int ml = wm + r * 16 + j + 8 * half;
        float g = sgate[ml];
        if (g != 0.f) {
          float val = g * c[r][ci][j];
          __hip_atomic_fetch_add(&Out[(size_t)stok[ml] * N + gn], val,
                                 __ATOMIC_RELAXED, __HIP_MEMORY_SCOPE_AGENT);
        }
      }
    }
  }
}

// ---------------- final residual add ----------------
__global__ void add_res_k(float* __restrict__ out, const float* __restrict__ h2, int n) {
  int i = blockIdx.x * blockDim.x + threadIdx.x;
  if (i < n) out[i] += h2[i];
}

// ============================ launch ============================
extern "C" void kernel_launch(void* const* d_in, const int* in_sizes, int n_in,
                              void* d_out, int out_size, void* d_ws, size_t ws_size,
                              hipStream_t stream) {
  const float* hs          = (const float*)d_in[0];
  const float* freqs       = (const float*)d_in[1];
  const float* attn_norm_w = (const float*)d_in[2];
  const float* moe_norm_w  = (const float*)d_in[3];
  const float* q_a_w       = (const float*)d_in[4];
  const float* q_a_norm_w  = (const float*)d_in[5];
  const float* q_b_w       = (const float*)d_in[6];
  const float* kv_a_w      = (const float*)d_in[7];
  const float* kv_a_norm_w = (const float*)d_in[8];
  const float* kv_b_w      = (const float*)d_in[9];
  const float* out_w       = (const float*)d_in[10];
  const float* router_w    = (const float*)d_in[11];
  const float* w1          = (const float*)d_in[12];
  const float* w3          = (const float*)d_in[13];
  const float* w2          = (const float*)d_in[14];
  const float* sw1         = (const float*)d_in[15];
  const float* sw3         = (const float*)d_in[16];
  const float* sw2         = (const float*)d_in[17];
  float* out = (float*)d_out;

  char* base = (char*)d_ws;
  size_t off = 0;
  auto alloc = [&](size_t bytes) { size_t o = off; off += (bytes + 255) & ~(size_t)255; return (void*)(base + o); };

  unsigned short* xb      = (unsigned short*)alloc((size_t)Tk * Dk * 2);
  unsigned short* qlat    = (unsigned short*)alloc((size_t)Tk * QLORA * 2);
  unsigned short* q       = (unsigned short*)alloc((size_t)Tk * Hk * 192 * 2);
  unsigned short* kva     = (unsigned short*)alloc((size_t)Tk * 576 * 2);
  unsigned short* kvb     = (unsigned short*)alloc((size_t)Tk * Hk * 256 * 2);
  unsigned short* attn    = (unsigned short*)alloc((size_t)Tk * Hk * 128 * 2);
  float*          h2      = (float*)alloc((size_t)Tk * Dk * 4);
  unsigned short* x2b     = (unsigned short*)alloc((size_t)Tk * Dk * 2);
  unsigned short* sharedh = (unsigned short*)alloc((size_t)Tk * SHMk * 2);
  unsigned short* moeh    = (unsigned short*)alloc((size_t)ROWS_TOT * Mk * 2);
  int*   topi   = (int*)alloc((size_t)Tk * 2 * 4);
  float* topv   = (float*)alloc((size_t)Tk * 2 * 4);
  int*   counts = (int*)alloc(Ek * 4);
  int*   seg    = (int*)alloc((Ek + 1) * 4);
  int*   pos    = (int*)alloc(Ek * 4);
  int*   tok    = (int*)alloc(ROWS_TOT * 4);
  float* gate   = (float*)alloc(ROWS_TOT * 4);
  (void)ws_size; (void)in_sizes; (void)n_in; (void)out_size;

  // ---- attention path ----
  rmsnorm_f32_k<<<Tk, 256, 0, stream>>>(hs, attn_norm_w, xb, Dk);
  gemm_k<1><<<dim3(Tk / 64, QLORA / 64), 128, 0, stream>>>(xb, Dk, q_a_w, qlat, nullptr, QLORA, Dk);
  rmsnorm_bf16_k<<<Tk, 256, 0, stream>>>(qlat, QLORA, QLORA, q_a_norm_w);
  gemm_k<1><<<dim3(Tk / 64, (Hk * 192) / 64), 128, 0, stream>>>(qlat, QLORA, q_b_w, q, nullptr, Hk * 192, QLORA);
  rope_k<<<(Tk * Hk * 32 + 255) / 256, 256, 0, stream>>>(q, Hk * 192, Hk, 192, NOPEk, freqs, Tk, Sk);
  gemm_k<1><<<dim3(Tk / 64, 576 / 64), 128, 0, stream>>>(xb, Dk, kv_a_w, kva, nullptr, 576, Dk);
  rmsnorm_bf16_k<<<Tk, 256, 0, stream>>>(kva, 576, KVLORA, kv_a_norm_w);
  rope_k<<<(Tk * 32 + 255) / 256, 256, 0, stream>>>(kva, 576, 1, 0, KVLORA, freqs, Tk, Sk);
  gemm_k<1><<<dim3(Tk / 64, (Hk * 256) / 64), 128, 0, stream>>>(kva, 576, kv_b_w, kvb, nullptr, Hk * 256, KVLORA);
  attention_k<<<dim3(Sk / 64, Hk, Bk), 128, 0, stream>>>(q, kvb, kva + KVLORA, attn, Sk);
  gemm_k<2><<<dim3(Tk / 64, Dk / 64), 128, 0, stream>>>(attn, Hk * 128, out_w, h2, hs, Dk, Hk * 128);

  // ---- MoE path ----
  rmsnorm_f32_k<<<Tk, 256, 0, stream>>>(h2, moe_norm_w, x2b, Dk);
  zero_counts_k<<<1, 32, 0, stream>>>(counts);
  router_k<<<Tk, 128, 0, stream>>>(x2b, router_w, topi, topv, counts);
  seg_k<<<1, 1, 0, stream>>>(counts, seg, pos);
  assign_init_k<<<(ROWS_TOT + 255) / 256, 256, 0, stream>>>(tok, gate);
  scatter_assign_k<<<(Tk + 255) / 256, 256, 0, stream>>>(topi, topv, seg, pos, tok, gate);

  // shared expert: writes d_out (covers every element, clears poison)
  ffn13_k<<<dim3(Tk / 64, SHMk / 64), 128, 0, stream>>>(x2b, Dk, sw1, sw3, 0, nullptr, nullptr,
                                                        sharedh, SHMk, SHMk, Dk);
  gemm_k<0><<<dim3(Tk / 64, Dk / 64), 128, 0, stream>>>(sharedh, SHMk, sw2, out, nullptr, Dk, SHMk);

  // routed experts: gathered FFN + gated scatter-accumulate into d_out
  ffn13_k<<<dim3(ROWS_TOT / 64, Mk / 64), 128, 0, stream>>>(x2b, Dk, w1, w3, (size_t)Mk * Dk,
                                                            tok, seg, moeh, Mk, Mk, Dk);
  moe_gemm2_k<<<dim3(ROWS_TOT / 64, Dk / 64), 128, 0, stream>>>(moeh, w2, (size_t)Dk * Mk,
                                                                tok, gate, seg, out, Dk, Mk);

  // ---- final residual ----
  add_res_k<<<(Tk * Dk + 255) / 256, 256, 0, stream>>>(out, h2, Tk * Dk);
}
